// ChebyNet_49830210568832
// MI455X (gfx1250) — matrix operand order repaired and checked by reading, not compiled, not run on hardware
//
#include <hip/hip_runtime.h>
#include <hip/hip_bf16.h>

// ---------------------------------------------------------------------------
// CDNA5 (gfx1250, wave32) WMMA-based MLP pipeline for the ChebyNet reference.
// bf16 inputs, f32 accumulation via v_wmma_f32_16x16x32_bf16.
// Global->LDS staging uses CDNA5 async copies (ASYNCcnt) with double buffering.
// ---------------------------------------------------------------------------

typedef __attribute__((ext_vector_type(16))) __bf16 v16bf;
typedef __attribute__((ext_vector_type(8)))  __bf16 v8bf;
typedef __attribute__((ext_vector_type(8)))  float  v8f;

#define TILE_M 128
#define TILE_N 128
#define TILE_K 64
#define LDSS   80   // padded LDS row stride in bf16 elems (160B: 16B aligned, bank-spreading)

// ---------------- f32 -> bf16 conversion ----------------
__global__ void k_cvt_bf16(const float* __restrict__ in, __bf16* __restrict__ out, int n) {
    int i = blockIdx.x * blockDim.x + threadIdx.x;
    if (i < n) out[i] = (__bf16)in[i];
}

// Low 32 bits of a generic pointer to a __shared__ object == LDS byte offset
// (ISA aperture rule: the LDS address is the pointer's low 32 bits).
__device__ __forceinline__ unsigned lds_off32(const void* p) {
    return (unsigned)(uintptr_t)p;
}

// ---------------- WMMA GEMM: C[M,Nout] = A[M,K](bf16) @ Bw[Nout,K]^T(bf16) + bias ----------------
// Bw is the weight in its natural [Hout, K] row-major layout (== B column-major), so
// A-tile and B-tile share the identical [row][k] LDS layout and fragment-load pattern.
__global__ __launch_bounds__(256) void k_gemm_bf16(
    const __bf16* __restrict__ A, const __bf16* __restrict__ Bw,
    const float* __restrict__ bias, float* __restrict__ C,
    int M, int K, int Nout, int relu)
{
    __shared__ __bf16 sA[2][TILE_M * LDSS];
    __shared__ __bf16 sB[2][TILE_N * LDSS];

    const int tid  = threadIdx.x;
    const int lane = tid & 31;
    const int wid  = tid >> 5;     // 8 waves
    const int wm   = wid >> 2;     // 0..1 : 64-row slab
    const int wn   = wid & 3;      // 0..3 : 32-col slab
    const int hsel = lane >> 4;    // lane half (0/1)
    const int l16  = lane & 15;

    const int bm = blockIdx.x * TILE_M;
    const int bn = blockIdx.y * TILE_N;

    v8f acc[4][2];
#pragma unroll
    for (int i = 0; i < 4; ++i)
#pragma unroll
        for (int j = 0; j < 2; ++j)
#pragma unroll
            for (int e = 0; e < 8; ++e) acc[i][j][e] = 0.0f;

    // Per-thread chunk mapping for cooperative tile loads:
    // 1024 chunks of 16B per 128x64 bf16 tile, 4 chunks per thread.
    // Clamp A rows to M-1: out-of-range rows only feed masked output rows,
    // so clamping is safe and keeps EXEC fully enabled (no divergence).
    const __bf16* gA[4];
    const __bf16* gB[4];
    int rr[4], cc[4];
#pragma unroll
    for (int it = 0; it < 4; ++it) {
        int idx = tid + it * 256;          // 0..1023
        int r   = idx >> 3;                // tile row 0..127
        int ck  = (idx & 7) << 3;          // bf16 col 0,8,..,56
        rr[it] = r; cc[it] = ck;
        int gr = bm + r; if (gr >= M) gr = M - 1;
        gA[it] = A  + (size_t)gr * K + ck;
        gB[it] = Bw + (size_t)(bn + r) * K + ck;   // Nout multiple of 128 -> in range
    }

    // Issue one tile's worth of async global->LDS copies (8 per thread).
    auto issue_tile = [&](int buf, int kt) {
#pragma unroll
        for (int it = 0; it < 4; ++it) {
            unsigned la = lds_off32(&sA[buf][rr[it] * LDSS + cc[it]]);
            unsigned lb = lds_off32(&sB[buf][rr[it] * LDSS + cc[it]]);
            asm volatile("global_load_async_to_lds_b128 %0, %1, off"
                         :: "v"(la), "v"(gA[it] + kt) : "memory");
            asm volatile("global_load_async_to_lds_b128 %0, %1, off"
                         :: "v"(lb), "v"(gB[it] + kt) : "memory");
        }
    };

    issue_tile(0, 0);
    int buf = 0;

    for (int kt = 0; kt < K; kt += TILE_K) {
        const bool more = (kt + TILE_K) < K;
        if (more) {
            issue_tile(buf ^ 1, kt + TILE_K);
            // async loads complete in order: <=8 outstanding means current tile landed
            asm volatile("s_wait_asynccnt 0x8" ::: "memory");
        } else {
            asm volatile("s_wait_asynccnt 0x0" ::: "memory");
        }
        __syncthreads();

        const __bf16* curA = sA[buf];
        const __bf16* curB = sB[buf];
#pragma unroll
        for (int ks = 0; ks < TILE_K; ks += 32) {
            v16bf afrag[4];
            v16bf bfrag[2];
            // A fragment (16x32 bf16): lane half picks k=8h+0..7 (VGPR0-3) and k=16+8h+0..7 (VGPR4-7)
#pragma unroll
            for (int i = 0; i < 4; ++i) {
                const __bf16* p = &curA[(wm * 64 + i * 16 + l16) * LDSS + ks + hsel * 8];
                v8bf lo = *(const v8bf*)p;
                v8bf hi = *(const v8bf*)(p + 16);
                afrag[i] = __builtin_shufflevector(lo, hi, 0,1,2,3,4,5,6,7,8,9,10,11,12,13,14,15);
            }
            // B fragment (32x16 bf16): lane n = l16; lane half picks k=8h+0..7 and k=16+8h+0..7 (same as A)
#pragma unroll
            for (int j = 0; j < 2; ++j) {
                const __bf16* p = &curB[(wn * 32 + j * 16 + l16) * LDSS + ks + hsel * 8];
                v8bf lo = *(const v8bf*)p;
                v8bf hi = *(const v8bf*)(p + 16);
                bfrag[j] = __builtin_shufflevector(lo, hi, 0,1,2,3,4,5,6,7,8,9,10,11,12,13,14,15);
            }
#pragma unroll
            for (int i = 0; i < 4; ++i)
#pragma unroll
                for (int j = 0; j < 2; ++j)
                    acc[i][j] = __builtin_amdgcn_wmma_f32_16x16x32_bf16(
                        false, afrag[i], false, bfrag[j],
                        (short)0, acc[i][j], false, false);
        }
        __syncthreads();   // all waves done reading `buf` before it is refilled
        buf ^= 1;
    }

    // ---- epilogue: bias (+optional ReLU), f32 store; C/D layout: VGPR j -> M=j+8*half ----
#pragma unroll
    for (int i = 0; i < 4; ++i) {
#pragma unroll
        for (int j = 0; j < 2; ++j) {
            int col = bn + wn * 32 + j * 16 + l16;
            float bv = bias ? bias[col] : 0.0f;
#pragma unroll
            for (int r8 = 0; r8 < 8; ++r8) {
                int row = bm + wm * 64 + i * 16 + hsel * 8 + r8;
                if (row < M) {
                    float v = acc[i][j][r8] + bv;
                    if (relu) v = fmaxf(v, 0.0f);
                    C[(size_t)row * Nout + col] = v;
                }
            }
        }
    }
}

// ---------------- BN: deterministic two-stage column reduction ----------------
#define BN_ROWS_PER_BLOCK 1024

__global__ void k_bn_partial(const float* __restrict__ h, int Nrows, int H,
                             float* __restrict__ sumPart, float* __restrict__ sqPart)
{
    int col = blockIdx.x * blockDim.x + threadIdx.x;       // < H
    int r0 = blockIdx.y * BN_ROWS_PER_BLOCK;
    int r1 = min(r0 + BN_ROWS_PER_BLOCK, Nrows);
    float s = 0.0f, sq = 0.0f;
    for (int r = r0; r < r1; ++r) {
        float v = h[(size_t)r * H + col];
        s += v; sq += v * v;
    }
    sumPart[(size_t)blockIdx.y * H + col] = s;
    sqPart [(size_t)blockIdx.y * H + col] = sq;
}

__global__ void k_bn_finalize(const float* __restrict__ sumPart, const float* __restrict__ sqPart,
                              int nchunks, int H, int Nrows,
                              const float* __restrict__ gamma, const float* __restrict__ beta,
                              float* __restrict__ scale, float* __restrict__ shift)
{
    int col = blockIdx.x * blockDim.x + threadIdx.x;
    if (col >= H) return;
    float s = 0.0f, sq = 0.0f;
    for (int c = 0; c < nchunks; ++c) {
        s  += sumPart[(size_t)c * H + col];
        sq += sqPart [(size_t)c * H + col];
    }
    float invN = 1.0f / (float)Nrows;
    float mean = s * invN;
    float var  = sq * invN - mean * mean;
    float sc   = gamma[col] * __frsqrt_rn(var + 1e-5f);
    scale[col] = sc;
    shift[col] = beta[col] - mean * sc;
}

// normalize + ReLU + convert to bf16 for next layer (4 elems / thread)
__global__ void k_bn_apply_relu_cvt(const float* __restrict__ h,
                                    const float* __restrict__ scale, const float* __restrict__ shift,
                                    __bf16* __restrict__ out, size_t total, int H)
{
    size_t i4 = ((size_t)blockIdx.x * blockDim.x + threadIdx.x) * 4;
    if (i4 >= total) return;
    float4 v = *(const float4*)(h + i4);
    int col = (int)(i4 % (size_t)H);        // H multiple of 4 -> no wrap across the 4
    float r0 = fmaxf(v.x * scale[col + 0] + shift[col + 0], 0.0f);
    float r1 = fmaxf(v.y * scale[col + 1] + shift[col + 1], 0.0f);
    float r2 = fmaxf(v.z * scale[col + 2] + shift[col + 2], 0.0f);
    float r3 = fmaxf(v.w * scale[col + 3] + shift[col + 3], 0.0f);
    out[i4 + 0] = (__bf16)r0;
    out[i4 + 1] = (__bf16)r1;
    out[i4 + 2] = (__bf16)r2;
    out[i4 + 3] = (__bf16)r3;
}

// ---------------- final layer: logits = h3 @ Wl2^T + bl2 ; log_softmax ----------------
// one wave per row; Wl2 [10,256] cached in LDS
__global__ __launch_bounds__(256) void k_final_logsoftmax(
    const float* __restrict__ h3, const float* __restrict__ W,
    const float* __restrict__ b, float* __restrict__ out, int M, int NC, int Kf)
{
    extern __shared__ float sW[];           // NC*Kf floats
    for (int i = threadIdx.x; i < NC * Kf; i += blockDim.x) sW[i] = W[i];
    __syncthreads();

    int wid  = threadIdx.x >> 5;
    int lane = threadIdx.x & 31;
    int row  = blockIdx.x * 8 + wid;
    if (row >= M) return;

    const float* hr = h3 + (size_t)row * Kf;
    float acc[10];
#pragma unroll
    for (int c = 0; c < 10; ++c) acc[c] = 0.0f;
    for (int i = lane; i < Kf; i += 32) {
        float hv = hr[i];
#pragma unroll
        for (int c = 0; c < 10; ++c) acc[c] += hv * sW[c * Kf + i];
    }
    // cross-lane sum (wave32)
#pragma unroll
    for (int c = 0; c < 10; ++c)
#pragma unroll
        for (int m = 16; m >= 1; m >>= 1)
            acc[c] += __shfl_xor(acc[c], m, 32);

    if (lane == 0) {
        float logits[10], mx = -3.402823e38f;
#pragma unroll
        for (int c = 0; c < 10; ++c) { logits[c] = acc[c] + b[c]; mx = fmaxf(mx, logits[c]); }
        float se = 0.0f;
#pragma unroll
        for (int c = 0; c < 10; ++c) se += __expf(logits[c] - mx);
        float lse = mx + __logf(se);
#pragma unroll
        for (int c = 0; c < 10; ++c) out[(size_t)row * 10 + c] = logits[c] - lse;
    }
}

// ---------------------------------------------------------------------------
extern "C" void kernel_launch(void* const* d_in, const int* in_sizes, int n_in,
                              void* d_out, int out_size, void* d_ws, size_t ws_size,
                              hipStream_t stream)
{
    const float* x   = (const float*)d_in[0];
    const float* W1  = (const float*)d_in[3];
    const float* b1  = (const float*)d_in[4];
    const float* g1  = (const float*)d_in[5];
    const float* be1 = (const float*)d_in[6];
    const float* W2  = (const float*)d_in[7];
    const float* b2  = (const float*)d_in[8];
    const float* g2  = (const float*)d_in[9];
    const float* be2 = (const float*)d_in[10];
    const float* Wl1 = (const float*)d_in[11];
    const float* bl1 = (const float*)d_in[12];
    const float* Wl2 = (const float*)d_in[13];
    const float* bl2 = (const float*)d_in[14];
    float* out = (float*)d_out;

    const int Din = 128;
    const int H   = in_sizes[4];            // 1024
    const int Hm  = in_sizes[12];           // 256
    const int NC  = in_sizes[14];           // 10
    const int N   = in_sizes[0] / Din;      // 50000

    // ---- workspace carve-out (aliased across layers) ----
    char* ws = (char*)d_ws;
    size_t off = 0;
    auto carve = [&](size_t bytes) -> void* {
        void* p = (void*)(ws + off);
        off += (bytes + 255) & ~(size_t)255;
        return p;
    };
    float*  bufA  = (float*) carve((size_t)N * H * 4);   // h1_f32 -> h2_f32 -> h3_f32
    __bf16* bufB  = (__bf16*)carve((size_t)N * H * 2);   // h1_bf16 -> h2_bf16
    __bf16* xb    = (__bf16*)carve((size_t)N * Din * 2);
    __bf16* W1b   = (__bf16*)carve((size_t)H * Din * 2);
    __bf16* W2b   = (__bf16*)carve((size_t)H * H * 2);
    __bf16* Wl1b  = (__bf16*)carve((size_t)Hm * H * 2);
    const int nchunks = (N + BN_ROWS_PER_BLOCK - 1) / BN_ROWS_PER_BLOCK;
    float* sumPart = (float*)carve((size_t)nchunks * H * 4);
    float* sqPart  = (float*)carve((size_t)nchunks * H * 4);
    float* bnScale = (float*)carve((size_t)H * 4);
    float* bnShift = (float*)carve((size_t)H * 4);
    (void)ws_size;

    auto cvt = [&](const float* src, __bf16* dst, int n) {
        k_cvt_bf16<<<(n + 255) / 256, 256, 0, stream>>>(src, dst, n);
    };
    cvt(x,   xb,   N * Din);
    cvt(W1,  W1b,  H * Din);
    cvt(W2,  W2b,  H * H);
    cvt(Wl1, Wl1b, Hm * H);

    dim3 blk(256);
    const int gM = (N + TILE_M - 1) / TILE_M;

    // ---- layer 1: [N,128] @ [128,1024]^T + b1 ----
    k_gemm_bf16<<<dim3(gM, H / TILE_N), blk, 0, stream>>>(xb, W1b, b1, bufA, N, Din, H, 0);
    // BN1 + ReLU -> bf16
    k_bn_partial<<<dim3(H / 256, nchunks), blk, 0, stream>>>(bufA, N, H, sumPart, sqPart);
    k_bn_finalize<<<dim3((H + 255) / 256), blk, 0, stream>>>(sumPart, sqPart, nchunks, H, N, g1, be1, bnScale, bnShift);
    {
        size_t total = (size_t)N * H;
        size_t nthr = total / 4;
        k_bn_apply_relu_cvt<<<dim3((unsigned)((nthr + 255) / 256)), blk, 0, stream>>>(bufA, bnScale, bnShift, bufB, total, H);
    }

    // ---- layer 2: [N,1024] @ [1024,1024]^T + b2 ----
    k_gemm_bf16<<<dim3(gM, H / TILE_N), blk, 0, stream>>>(bufB, W2b, b2, bufA, N, H, H, 0);
    k_bn_partial<<<dim3(H / 256, nchunks), blk, 0, stream>>>(bufA, N, H, sumPart, sqPart);
    k_bn_finalize<<<dim3((H + 255) / 256), blk, 0, stream>>>(sumPart, sqPart, nchunks, H, N, g2, be2, bnScale, bnShift);
    {
        size_t total = (size_t)N * H;
        size_t nthr = total / 4;
        k_bn_apply_relu_cvt<<<dim3((unsigned)((nthr + 255) / 256)), blk, 0, stream>>>(bufA, bnScale, bnShift, bufB, total, H);
    }

    // ---- layer 3: [N,1024] @ [1024,256]^T + bl1, fused ReLU -> f32 (overwrites bufA) ----
    k_gemm_bf16<<<dim3(gM, Hm / TILE_N), blk, 0, stream>>>(bufB, Wl1b, bl1, bufA, N, H, Hm, 1);

    // ---- layer 4 + log_softmax ----
    size_t smem = (size_t)NC * Hm * sizeof(float);
    k_final_logsoftmax<<<dim3((N + 7) / 8), blk, smem, stream>>>(bufA, Wl2, bl2, out, N, NC, Hm);
    (void)out_size;
}
